// Block_45165876085628
// MI455X (gfx1250) — matrix operand order repaired and checked by reading, not compiled, not run on hardware
//
#include <hip/hip_runtime.h>

#define DIMD   768
#define NHEADS 12
#define HDIM   64
#define SLEN   197
#define BATCH  64
#define MROWS  (BATCH * SLEN)   // 12608
#define HIDDEN 3072

typedef unsigned short u16;
typedef __attribute__((ext_vector_type(4)))  u16          v4us;
typedef __attribute__((ext_vector_type(8)))  u16          v8us;
typedef __attribute__((ext_vector_type(16))) __bf16       v16bf;
typedef __attribute__((ext_vector_type(8)))  float        v8f;
typedef __attribute__((ext_vector_type(4)))  unsigned int v4u;
typedef __attribute__((ext_vector_type(4)))  int          v4i;
typedef __attribute__((ext_vector_type(8)))  int          v8i;

union BFu { v16bf bf; v8us h[2]; };

__device__ __forceinline__ u16 f2bf(float f) {
    union { float f; unsigned u; } c; c.f = f;
    unsigned u = c.u;
    return (u16)((u + 0x7FFFu + ((u >> 16) & 1u)) >> 16);  // RNE
}

// A-fragment: element i -> k = half*16 + i; p at row*stride + half*16
__device__ __forceinline__ v16bf ldfragA(const u16* p) {
    BFu r; r.h[0] = *(const v8us*)p; r.h[1] = *(const v8us*)(p + 8); return r.bf;
}
// B-fragment: element i -> k = half*16 + i; p at n*stride + half*16
__device__ __forceinline__ v16bf ldfragB(const u16* p) {
    BFu r; r.h[0] = *(const v8us*)p; r.h[1] = *(const v8us*)(p + 8); return r.bf;
}

// ---------------------------------------------------------------------------
// Tensor Data Mover: 2D bf16 tile (global, row stride stride_e elements) ->
// LDS with row padding. tensor_d1 < tile_d1 -> hardware zero-fill of OOB rows.
// ---------------------------------------------------------------------------
__device__ __forceinline__ void tdm_load_2d(unsigned lds_addr, const void* gptr,
                                            unsigned tile_d0, unsigned tile_d1,
                                            unsigned tensor_d0, unsigned tensor_d1,
                                            unsigned stride_e,
                                            unsigned pad_iv, unsigned pad_am) {
    unsigned long long ga = (unsigned long long)(uintptr_t)gptr;
    v4u g0;
    g0[0] = 1u;                                       // count=1, user descriptor
    g0[1] = lds_addr;                                 // LDS byte address
    g0[2] = (unsigned)(ga & 0xFFFFFFFFu);             // global_addr[31:0]
    g0[3] = (unsigned)((ga >> 32) & 0x01FFFFFFu) | (2u << 30);  // addr[56:32] | type=2
    v8i g1;
    g1[0] = (int)((1u << 16) |                        // data_size = 2 bytes
                  (1u << 20) |                        // pad_enable
                  (pad_iv << 22) | (pad_am << 25));
    g1[1] = (int)((tensor_d0 & 0xFFFFu) << 16);       // tensor_dim0[15:0]
    g1[2] = (int)(((tensor_d0 >> 16) & 0xFFFFu) | ((tensor_d1 & 0xFFFFu) << 16));
    g1[3] = (int)(((tensor_d1 >> 16) & 0xFFFFu) | ((tile_d0 & 0xFFFFu) << 16));
    g1[4] = (int)(tile_d1 & 0xFFFFu);                 // tile_dim1 (tile_dim2=0)
    g1[5] = (int)stride_e;                            // tensor_dim0_stride[31:0]
    g1[6] = 0;
    g1[7] = 0;
    v4i gz = {0, 0, 0, 0};
#if defined(__clang_major__) && (__clang_major__ >= 23)
    v8i gz8 = {0, 0, 0, 0, 0, 0, 0, 0};
    __builtin_amdgcn_tensor_load_to_lds(g0, g1, gz, gz, gz8, 0);
#else
    __builtin_amdgcn_tensor_load_to_lds(g0, g1, gz, gz, 0);
#endif
}

// ---------------------------------------------------------------------------
// Weight prep: W[K,N] f32 -> Wt[N,K] bf16 (32x32 LDS-tiled transpose).
// ---------------------------------------------------------------------------
__global__ __launch_bounds__(256) void wtrans_kernel(const float* __restrict__ W,
                                                     u16* __restrict__ Wt,
                                                     int K, int N) {
    __shared__ float tile[32][33];
    const int kb = blockIdx.y * 32, nb = blockIdx.x * 32;
    const int tx = threadIdx.x & 31, ty = threadIdx.x >> 5;
#pragma unroll
    for (int i = ty; i < 32; i += 8)
        tile[i][tx] = W[(size_t)(kb + i) * N + nb + tx];
    __syncthreads();
#pragma unroll
    for (int i = ty; i < 32; i += 8)
        Wt[(size_t)(nb + i) * K + kb + tx] = f2bf(tile[tx][i]);
}

// ---------------------------------------------------------------------------
// Elementwise f32 -> bf16 (float4 / v4us granularity).
// ---------------------------------------------------------------------------
__global__ __launch_bounds__(256) void cvt_bf16_kernel(const float* __restrict__ src,
                                                       u16* __restrict__ dst, int n4) {
    int idx = blockIdx.x * 256 + threadIdx.x;
    if (idx < n4) {
        float4 a = ((const float4*)src)[idx];
        v4us st = {f2bf(a.x), f2bf(a.y), f2bf(a.z), f2bf(a.w)};
        ((v4us*)dst)[idx] = st;
    }
}

// ---------------------------------------------------------------------------
// Tiled GEMM: C[M,N] = Ab[M,K](bf16) * Wt[N,K](bf16)^T + bias, optional GELU.
// 256 thr (8 waves), 128x128 tile, K step 32. A and B tiles double-buffered,
// both staged by TDM (wave 0 issues; barrier + s_wait_tensorcnt pipelines).
// ---------------------------------------------------------------------------
template <int EPI, int OUTBF>
__global__ __launch_bounds__(256) void gemm_bias(const u16* __restrict__ Ab,
                                                 const u16* __restrict__ Wt,
                                                 const float* __restrict__ bias,
                                                 void* __restrict__ Cout,
                                                 int M, int N, int K) {
    __shared__ u16 As[2][128 * 40];   // [row][k], stride 40 halves (TDM-padded)
    __shared__ u16 Bs[2][128 * 40];   // [n][k],  stride 40 halves (TDM-padded)

    const int t    = threadIdx.x;
    const int lane = t & 31, w = t >> 5;
    const int wm   = w >> 2, wn = w & 3;
    const int hl   = lane >> 4, ln = lane & 15;
    const int row0 = blockIdx.y * 128, col0 = blockIdx.x * 128;
    const int nk   = K >> 5;
    const unsigned mrem = (unsigned)(M - row0);   // ragged M edge -> TDM zero-fill

    if (w == 0) {
        tdm_load_2d((unsigned)(uintptr_t)&As[0][0], Ab + (size_t)row0 * K,
                    32, 128, 32, mrem, (unsigned)K, 3, 3);
        tdm_load_2d((unsigned)(uintptr_t)&Bs[0][0], Wt + (size_t)col0 * K,
                    32, 128, 32, 128, (unsigned)K, 3, 3);
    }

    const v8f z = {0.f, 0.f, 0.f, 0.f, 0.f, 0.f, 0.f, 0.f};
    v8f acc[4][2];
#pragma unroll
    for (int i = 0; i < 4; i++)
#pragma unroll
        for (int j = 0; j < 2; j++) acc[i][j] = z;

    for (int kt = 0; kt < nk; kt++) {
        const int buf = kt & 1;
        const int k0 = kt << 5;
        __builtin_amdgcn_s_wait_tensorcnt(0);   // current buf tiles ready (wave0)
        __syncthreads();
        if (w == 0 && kt + 1 < nk) {            // async-stage next tiles
            tdm_load_2d((unsigned)(uintptr_t)&As[buf ^ 1][0],
                        Ab + (size_t)row0 * K + k0 + 32,
                        32, 128, 32, mrem, (unsigned)K, 3, 3);
            tdm_load_2d((unsigned)(uintptr_t)&Bs[buf ^ 1][0],
                        Wt + (size_t)col0 * K + k0 + 32,
                        32, 128, 32, 128, (unsigned)K, 3, 3);
        }

        v16bf af[4], bfr[2];
#pragma unroll
        for (int am = 0; am < 4; am++)
            af[am] = ldfragA(&As[buf][(wm * 64 + am * 16 + ln) * 40 + hl * 16]);
#pragma unroll
        for (int bn = 0; bn < 2; bn++)
            bfr[bn] = ldfragB(&Bs[buf][(wn * 32 + bn * 16 + ln) * 40 + hl * 16]);
#pragma unroll
        for (int am = 0; am < 4; am++)
#pragma unroll
            for (int bn = 0; bn < 2; bn++)
                acc[am][bn] = __builtin_amdgcn_wmma_f32_16x16x32_bf16(
                    false, af[am], false, bfr[bn], (short)0, acc[am][bn], false, false);
    }

    float bv[2];
#pragma unroll
    for (int bn = 0; bn < 2; bn++) bv[bn] = bias[col0 + wn * 32 + bn * 16 + ln];

#pragma unroll
    for (int am = 0; am < 4; am++)
#pragma unroll
        for (int bn = 0; bn < 2; bn++)
#pragma unroll
            for (int r = 0; r < 8; r++) {
                int m = row0 + wm * 64 + am * 16 + hl * 8 + r;
                if (m < M) {
                    float v = acc[am][bn][r] + bv[bn];
                    if (EPI == 1) v = 0.5f * v * (1.0f + erff(v * 0.70710678118654752f));
                    size_t o = (size_t)m * N + col0 + wn * 32 + bn * 16 + ln;
                    if (OUTBF) ((u16*)Cout)[o] = f2bf(v);
                    else       ((float*)Cout)[o] = v;
                }
            }
}

// ---------------------------------------------------------------------------
// Flash attention: one workgroup (128 thr, 4 waves) per (b, h); bf16 in/out.
// K staged by TDM (OOB rows auto-zeroed); per-wave Q blocks staged by TDM.
// ---------------------------------------------------------------------------
__global__ __launch_bounds__(128) void attn_kernel(const u16* __restrict__ Qg,
                                                   const u16* __restrict__ Kg,
                                                   const u16* __restrict__ Vg,
                                                   u16* __restrict__ Cg) {
    __shared__ u16 Ks[208 * 72];    // [t][hd], rows 197..207 zero (TDM OOB)
    __shared__ u16 Vt[64 * 232];    // [hd][s], cols 197..223 zeroed
    __shared__ u16 Qw[4][16 * 72];  // per-wave Q block
    __shared__ u16 Pw[4][16 * 40];  // per-wave P block (16 x 32)

    const int b = blockIdx.y, hh = blockIdx.x;
    const int t = threadIdx.x, lane = t & 31, w = t >> 5;
    const int hl = lane >> 4, ln = lane & 15;
    const size_t base = ((size_t)b * SLEN) * DIMD + (size_t)hh * HDIM;

    if (w == 0)
        tdm_load_2d((unsigned)(uintptr_t)&Ks[0], Kg + base,
                    64, 208, 64, SLEN, DIMD, 4, 3);
    for (int idx = t; idx < 2048; idx += 128) {
        int d = idx >> 5, s0 = 192 + (idx & 31);
        if (s0 >= SLEN) Vt[d * 232 + s0] = (u16)0;
    }
    for (int idx = t; idx < SLEN * 16; idx += 128) {
        int s = idx >> 4, c = (idx & 15) << 2;
        v4us vv = *(const v4us*)(Vg + base + (size_t)s * DIMD + c);
        Vt[(c + 0) * 232 + s] = vv[0];
        Vt[(c + 1) * 232 + s] = vv[1];
        Vt[(c + 2) * 232 + s] = vv[2];
        Vt[(c + 3) * 232 + s] = vv[3];
    }
    __builtin_amdgcn_s_wait_tensorcnt(0);
    __syncthreads();

    u16* qw = Qw[w];
    u16* pw = Pw[w];
    const v8f z = {0.f, 0.f, 0.f, 0.f, 0.f, 0.f, 0.f, 0.f};

    for (int rb = w; rb < 13; rb += 4) {
        tdm_load_2d((unsigned)(uintptr_t)qw, Qg + base + (size_t)rb * 16 * DIMD,
                    64, 16, 64, (unsigned)(SLEN - rb * 16), DIMD, 4, 3);
        __builtin_amdgcn_s_wait_tensorcnt(0);

        const u16* qp = &qw[ln * 72 + hl * 16];
        v16bf qlo = ldfragA(qp);
        v16bf qhi = ldfragA(qp + 32);

        v8f o[4];
        float mrow[8], lrow[8];
#pragma unroll
        for (int dt = 0; dt < 4; dt++) o[dt] = z;
#pragma unroll
        for (int r = 0; r < 8; r++) { mrow[r] = -1e30f; lrow[r] = 0.f; }

        for (int jp = 0; jp < 7; jp++) {
            int jt0 = jp * 2, jt1 = jp * 2 + 1;
            v8f s0 = z, s1 = z;
            {
                int tc = jt0 * 16 + ln;
                const u16* kp = &Ks[tc * 72 + hl * 16];
                v16bf b0 = ldfragB(kp);
                v16bf b1 = ldfragB(kp + 32);
                s0 = __builtin_amdgcn_wmma_f32_16x16x32_bf16(false, qlo, false, b0,
                                                             (short)0, s0, false, false);
                s0 = __builtin_amdgcn_wmma_f32_16x16x32_bf16(false, qhi, false, b1,
                                                             (short)0, s0, false, false);
                s0 = s0 * 0.125f;
                if (tc >= SLEN) {
#pragma unroll
                    for (int r = 0; r < 8; r++) s0[r] = -1e30f;
                }
            }
            if (jt1 < 13) {
                int tc = jt1 * 16 + ln;
                const u16* kp = &Ks[tc * 72 + hl * 16];
                v16bf b0 = ldfragB(kp);
                v16bf b1 = ldfragB(kp + 32);
                s1 = __builtin_amdgcn_wmma_f32_16x16x32_bf16(false, qlo, false, b0,
                                                             (short)0, s1, false, false);
                s1 = __builtin_amdgcn_wmma_f32_16x16x32_bf16(false, qhi, false, b1,
                                                             (short)0, s1, false, false);
                s1 = s1 * 0.125f;
                if (tc >= SLEN) {
#pragma unroll
                    for (int r = 0; r < 8; r++) s1[r] = -1e30f;
                }
            } else {
#pragma unroll
                for (int r = 0; r < 8; r++) s1[r] = -1e30f;
            }

            float fs[8];
#pragma unroll
            for (int r = 0; r < 8; r++) {
                float mx = fmaxf(s0[r], s1[r]);
                mx = fmaxf(mx, __shfl_xor(mx, 1, 32));
                mx = fmaxf(mx, __shfl_xor(mx, 2, 32));
                mx = fmaxf(mx, __shfl_xor(mx, 4, 32));
                mx = fmaxf(mx, __shfl_xor(mx, 8, 32));
                float mnew = fmaxf(mrow[r], mx);
                float p0 = __expf(s0[r] - mnew);
                float p1 = __expf(s1[r] - mnew);
                float rs = p0 + p1;
                rs += __shfl_xor(rs, 1, 32);
                rs += __shfl_xor(rs, 2, 32);
                rs += __shfl_xor(rs, 4, 32);
                rs += __shfl_xor(rs, 8, 32);
                float f = __expf(mrow[r] - mnew);
                lrow[r] = lrow[r] * f + rs;
                mrow[r] = mnew;
                fs[r] = f;
                int pr = hl * 8 + r;
                pw[pr * 40 + ln]      = f2bf(p0);
                pw[pr * 40 + 16 + ln] = f2bf(p1);
            }
#pragma unroll
            for (int dt = 0; dt < 4; dt++)
#pragma unroll
                for (int r = 0; r < 8; r++) o[dt][r] *= fs[r];

            const u16* pp = &pw[ln * 40 + hl * 16];
            v16bf pf = ldfragA(pp);
#pragma unroll
            for (int dt = 0; dt < 4; dt++) {
                const u16* vp = &Vt[(dt * 16 + ln) * 232 + jp * 32 + hl * 16];
                v16bf vf = ldfragB(vp);
                o[dt] = __builtin_amdgcn_wmma_f32_16x16x32_bf16(false, pf, false, vf,
                                                                (short)0, o[dt], false, false);
            }
        }

#pragma unroll
        for (int r = 0; r < 8; r++) {
            int s = rb * 16 + hl * 8 + r;
            if (s < SLEN) {
                float inv = 1.0f / lrow[r];
#pragma unroll
                for (int dt = 0; dt < 4; dt++)
                    Cg[base + (size_t)s * DIMD + dt * 16 + ln] = f2bf(o[dt][r] * inv);
            }
        }
    }
}

// ---------------------------------------------------------------------------
// out = xr + LayerNorm(y)*g + b ; optionally also emit bf16 copy of out.
// ---------------------------------------------------------------------------
template <int WB>
__global__ __launch_bounds__(256) void add_ln_kernel(const float* __restrict__ xr,
                                                     const float* __restrict__ y,
                                                     const float* __restrict__ g,
                                                     const float* __restrict__ be,
                                                     float* __restrict__ out,
                                                     u16* __restrict__ outb) {
    __shared__ float ss[8], ss2[8];
    const int row = blockIdx.x;
    const size_t off = (size_t)row * DIMD;
    const int t = threadIdx.x, lane = t & 31, w = t >> 5;

    float v[3], s = 0.f, s2 = 0.f;
#pragma unroll
    for (int i = 0; i < 3; i++) {
        v[i] = y[off + t + i * 256];
        s += v[i];
        s2 += v[i] * v[i];
    }
#pragma unroll
    for (int m = 16; m >= 1; m >>= 1) {
        s  += __shfl_xor(s,  m, 32);
        s2 += __shfl_xor(s2, m, 32);
    }
    if (lane == 0) { ss[w] = s; ss2[w] = s2; }
    __syncthreads();
    float tot = 0.f, tot2 = 0.f;
#pragma unroll
    for (int i = 0; i < 8; i++) { tot += ss[i]; tot2 += ss2[i]; }
    const float mu = tot * (1.0f / DIMD);
    const float var = tot2 * (1.0f / DIMD) - mu * mu;
    const float rstd = rsqrtf(var + 1e-6f);
#pragma unroll
    for (int i = 0; i < 3; i++) {
        int c = t + i * 256;
        float res = xr[off + c] + (v[i] - mu) * rstd * g[c] + be[c];
        out[off + c] = res;
        if (WB) outb[off + c] = f2bf(res);
    }
}

// ---------------------------------------------------------------------------
extern "C" void kernel_launch(void* const* d_in, const int* in_sizes, int n_in,
                              void* d_out, int out_size, void* d_ws, size_t ws_size,
                              hipStream_t stream) {
    const float* x  = (const float*)d_in[0];
    const float* Wq = (const float*)d_in[1];  const float* bq = (const float*)d_in[2];
    const float* Wk = (const float*)d_in[3];  const float* bk = (const float*)d_in[4];
    const float* Wv = (const float*)d_in[5];  const float* bv = (const float*)d_in[6];
    const float* Wo = (const float*)d_in[7];  const float* bo = (const float*)d_in[8];
    const float* W1 = (const float*)d_in[9];  const float* b1 = (const float*)d_in[10];
    const float* W2 = (const float*)d_in[11]; const float* b2 = (const float*)d_in[12];
    const float* g1 = (const float*)d_in[13]; const float* be1 = (const float*)d_in[14];
    const float* g2 = (const float*)d_in[15]; const float* be2 = (const float*)d_in[16];

    const int M = MROWS;
    const size_t MD = (size_t)M * DIMD;                 // 9,682,944
    float* ws = (float*)d_ws;

    // bf16 region (u16 units)
    u16* u = (u16*)ws;
    u16* qb   = u;                 // [M,D]
    u16* kb   = qb + MD;
    u16* vb   = kb + MD;
    u16* xb   = vb + MD;           // bf16(x)
    u16* ctxb = xb + MD;           // attention output (bf16)
    u16* x1b  = ctxb + MD;         // bf16(x1)
    u16* hb   = x1b + MD;          // [M,3072] bf16 (GELU output)
    // f32 region (10*MD u16 consumed above = 5*MD floats)
    float* attn_out = ws + 5 * MD;
    float* x1       = attn_out + MD;
    float* mlp_out  = x1 + MD;
    // pre-transposed bf16 weights
    u16* wts = (u16*)(mlp_out + MD);
    const size_t WSQ = (size_t)DIMD * DIMD;             // 589824
    const size_t WSH = (size_t)DIMD * HIDDEN;           // 2359296
    u16* wtq = wts;
    u16* wtk = wtq + WSQ;
    u16* wtv = wtk + WSQ;
    u16* wto = wtv + WSQ;
    u16* wt1 = wto + WSQ;   // [3072][768]
    u16* wt2 = wt1 + WSH;   // [768][3072]

    // weight prep: f32 [K,N] -> bf16 [N,K]
    wtrans_kernel<<<dim3(DIMD / 32, DIMD / 32), 256, 0, stream>>>(Wq, wtq, DIMD, DIMD);
    wtrans_kernel<<<dim3(DIMD / 32, DIMD / 32), 256, 0, stream>>>(Wk, wtk, DIMD, DIMD);
    wtrans_kernel<<<dim3(DIMD / 32, DIMD / 32), 256, 0, stream>>>(Wv, wtv, DIMD, DIMD);
    wtrans_kernel<<<dim3(DIMD / 32, DIMD / 32), 256, 0, stream>>>(Wo, wto, DIMD, DIMD);
    wtrans_kernel<<<dim3(HIDDEN / 32, DIMD / 32), 256, 0, stream>>>(W1, wt1, DIMD, HIDDEN);
    wtrans_kernel<<<dim3(DIMD / 32, HIDDEN / 32), 256, 0, stream>>>(W2, wt2, HIDDEN, DIMD);

    // activation prep: bf16(x)
    const int n4 = (int)(MD / 4);
    cvt_bf16_kernel<<<(n4 + 255) / 256, 256, 0, stream>>>(x, xb, n4);

    dim3 g6(6, (M + 127) / 128);    // N=768
    dim3 g24(24, (M + 127) / 128);  // N=3072

    gemm_bias<0, 1><<<g6, 256, 0, stream>>>(xb, wtq, bq, qb, M, DIMD, DIMD);
    gemm_bias<0, 1><<<g6, 256, 0, stream>>>(xb, wtk, bk, kb, M, DIMD, DIMD);
    gemm_bias<0, 1><<<g6, 256, 0, stream>>>(xb, wtv, bv, vb, M, DIMD, DIMD);

    attn_kernel<<<dim3(NHEADS, BATCH), 128, 0, stream>>>(qb, kb, vb, ctxb);

    gemm_bias<0, 0><<<g6, 256, 0, stream>>>(ctxb, wto, bo, attn_out, M, DIMD, DIMD);
    add_ln_kernel<1><<<M, 256, 0, stream>>>(x, attn_out, g1, be1, x1, x1b);

    gemm_bias<1, 1><<<g24, 256, 0, stream>>>(x1b, wt1, b1, hb, M, HIDDEN, DIMD);
    gemm_bias<0, 0><<<g6, 256, 0, stream>>>(hb, wt2, b2, mlp_out, M, DIMD, HIDDEN);
    add_ln_kernel<0><<<M, 256, 0, stream>>>(x1, mlp_out, g2, be2, (float*)d_out, nullptr);
}
